// Quantizer_60498909331651
// MI455X (gfx1250) — matrix operand order repaired and checked by reading, not compiled, not run on hardware
//
#include <hip/hip_runtime.h>

// RVQ: L=8 layers, G=2 groups, K=1024 codes, D=256 dims
// xin [B=16, C=512, T=4096] f32; codebooks [L,G,K,D] f32.
// Outputs (concat): quantized_out [B,C,T] f32, total_loss scalar, indices [L,G,N] (as float).

#define L_LAYERS 8
#define G_GRP    2
#define K_CODES  1024
#define D_DIM    256
#define B_BATCH  16
#define T_LEN    4096
#define C_CH     (G_GRP * D_DIM)            // 512
#define N_VEC    (B_BATCH * T_LEN)          // 65536
#define QN       ((size_t)B_BATCH * C_CH * T_LEN)  // 33554432

typedef __attribute__((ext_vector_type(16))) __bf16 v16bf;
typedef __attribute__((ext_vector_type(8)))  float  v8f;

__device__ __forceinline__ unsigned short f2bf(float f) {
  unsigned int u = __float_as_uint(f);
  u += 0x7FFFu + ((u >> 16) & 1u);   // round-to-nearest-even
  return (unsigned short)(u >> 16);
}

// ---------------- codebook prep: f32 -> bf16 + per-code squared norms ----------------
__global__ void prep_codebook(const float* __restrict__ cb,
                              unsigned short* __restrict__ cbbf,
                              float* __restrict__ norms) {
  int row = blockIdx.x * blockDim.x + threadIdx.x;   // (l,g,k) flattened
  if (row >= L_LAYERS * G_GRP * K_CODES) return;
  const float* src = cb + (size_t)row * D_DIM;
  unsigned short* dst = cbbf + (size_t)row * D_DIM;
  float s = 0.f;
  #pragma unroll 4
  for (int d = 0; d < D_DIM; ++d) {
    float v = src[d];
    s += v * v;
    dst[d] = f2bf(v);
  }
  norms[row] = s;
}

// ---------------- residual init: residual = xin ----------------
__global__ void copy_kernel(const float4* __restrict__ in, float4* __restrict__ out, size_t n4) {
  size_t i = (size_t)blockIdx.x * blockDim.x + threadIdx.x;
  size_t stride = (size_t)gridDim.x * blockDim.x;
  for (; i < n4; i += stride) out[i] = in[i];
}

// ---------------- fused per-layer kernel: argmin (WMMA) + gather + residual update + SSE ----
// Block = 64 threads (2 wave32). Block owns 64 rows; each wave owns 32 rows as TWO 16-row
// A tiles, so every B fragment loaded from the (L2-resident) bf16 codebook feeds two WMMAs.
__launch_bounds__(64)
__global__ void rvq_layer(float* __restrict__ R,                 // residual [B,C,T], updated in place
                          const float* __restrict__ cb32,       // [L,G,K,D] f32
                          const unsigned short* __restrict__ cbbf, // [L,G,K,D] bf16
                          const float* __restrict__ norms,      // [L,G,K]
                          float* __restrict__ idx_out_layer,    // [G,N] float-encoded, this layer
                          float* __restrict__ loss_acc,
                          int layer) {
  __shared__ unsigned short As[64][264];   // 64 rows x 256 d (+8 pad), bf16
  __shared__ float normS[K_CODES];
  __shared__ int   idxS[64];

  const int tid = threadIdx.x;             // 0..63
  const int g   = blockIdx.y;
  const int n0  = blockIdx.x * 64;         // 64 rows per block, all within one b (64 | 4096)
  const int b   = n0 >> 12;                // n = b*T + t
  const int t0  = n0 & (T_LEN - 1);
  float* Rblk = R + ((size_t)b * C_CH + (size_t)g * D_DIM) * T_LEN + t0;  // element (d,row): +d*T+row

  // ---- stage A tile: global f32 (d-major, stride T) -> LDS bf16 row-major [row][d] ----
  {
    const int row = tid;                   // lanes span 64 consecutive t -> coalesced
    #pragma unroll 4
    for (int d = 0; d < D_DIM; ++d)
      As[row][d] = f2bf(Rblk[(size_t)d * T_LEN + row]);
  }
  // ---- stage code norms ----
  const float* nrm = norms + ((size_t)layer * G_GRP + g) * K_CODES;
  for (int i = tid; i < K_CODES; i += 64) normS[i] = nrm[i];
  __syncthreads();

  const int wave  = tid >> 5;              // 2 waves, each owns 32 rows (two 16-row tiles)
  const int lane  = tid & 31;
  const int half  = lane >> 4;
  const int m     = lane & 15;
  const int rbase = wave * 32;

  // ---- preload A fragments for both 16-row tiles (full D=256 each) ----
  // A k order used here (contiguous, not AMD's table order; B matches): lanes<16 row=m K={0..15}; lanes>=16 row=m K={16..31}
  v16bf aF0[8], aF1[8];
  #pragma unroll
  for (int c = 0; c < 8; ++c) {
    const int dc = c * 32;
    union { v16bf v; uint4 q[2]; } u0, u1;
    u0.q[0] = *(const uint4*)&As[rbase + m][dc + 16 * half];
    u0.q[1] = *(const uint4*)&As[rbase + m][dc + 16 * half + 8];
    u1.q[0] = *(const uint4*)&As[rbase + 16 + m][dc + 16 * half];
    u1.q[1] = *(const uint4*)&As[rbase + 16 + m][dc + 16 * half + 8];
    aF0[c] = u0.v;
    aF1[c] = u1.v;
  }

  const unsigned short* cbl = cbbf + ((size_t)layer * G_GRP + g) * K_CODES * D_DIM;

  float best0[8], best1[8];
  int   bestk0[8], bestk1[8];
  #pragma unroll
  for (int v = 0; v < 8; ++v) {
    best0[v] = 3.4e38f; bestk0[v] = 0;
    best1[v] = 3.4e38f; bestk1[v] = 0;
  }

  // ---- sweep all 1024 codes, 16 at a time; score = ||e||^2 - 2 x.e ----
  for (int kt = 0; kt < K_CODES; kt += 16) {
    v8f acc0 = {}, acc1 = {};
    // B k order used here (same contiguous order as A): lane holds 16 consecutive K (=d) values of code column m,
    // K offset 16*half -> one contiguous 32-byte chunk of the bf16 codebook row.
    const unsigned short* brow = cbl + (size_t)(kt + m) * D_DIM + 16 * half;
    #pragma unroll
    for (int c = 0; c < 8; ++c) {
      union { v16bf v; uint4 q[2]; } u;
      const uint4* bp = (const uint4*)(brow + c * 32);
      u.q[0] = bp[0];
      u.q[1] = bp[1];
      acc0 = __builtin_amdgcn_wmma_f32_16x16x32_bf16(false, aF0[c], false, u.v,
                                                     (short)0, acc0, false, false);
      acc1 = __builtin_amdgcn_wmma_f32_16x16x32_bf16(false, aF1[c], false, u.v,
                                                     (short)0, acc1, false, false);
    }
    // C/D layout: VGPR v -> row v + 8*half, column (code) = kt + m
    const float nk = normS[kt + m];
    const int   kk = kt + m;
    #pragma unroll
    for (int v = 0; v < 8; ++v) {
      float s0 = nk - 2.0f * acc0[v];
      if (s0 < best0[v]) { best0[v] = s0; bestk0[v] = kk; }
      float s1 = nk - 2.0f * acc1[v];
      if (s1 < best1[v]) { best1[v] = s1; bestk1[v] = kk; }
    }
  }

  // ---- argmin across the 16 code-columns (lanes with same half) ----
  #pragma unroll
  for (int off = 1; off < 16; off <<= 1) {
    #pragma unroll
    for (int v = 0; v < 8; ++v) {
      float os0 = __shfl_xor(best0[v], off, 32);
      int   ok0 = __shfl_xor(bestk0[v], off, 32);
      if (os0 < best0[v] || (os0 == best0[v] && ok0 < bestk0[v])) { best0[v] = os0; bestk0[v] = ok0; }
      float os1 = __shfl_xor(best1[v], off, 32);
      int   ok1 = __shfl_xor(bestk1[v], off, 32);
      if (os1 < best1[v] || (os1 == best1[v] && ok1 < bestk1[v])) { best1[v] = os1; bestk1[v] = ok1; }
    }
  }
  if (m == 0) {
    float* idx_out = idx_out_layer + (size_t)g * N_VEC;
    #pragma unroll
    for (int v = 0; v < 8; ++v) {
      int row0 = rbase + half * 8 + v;
      int row1 = rbase + 16 + half * 8 + v;
      idxS[row0] = bestk0[v];
      idxS[row1] = bestk1[v];
      idx_out[n0 + row0] = (float)bestk0[v];   // jnp argmin -> int; emit as float value
      idx_out[n0 + row1] = (float)bestk1[v];
    }
  }
  __syncthreads();

  // ---- gather winning code (f32), update residual in place, accumulate SSE ----
  // new_r = x - e_k ; per-layer loss contribution = 1.25 * mean(new_r^2)
  const float* cbl32 = cb32 + ((size_t)layer * G_GRP + g) * K_CODES * D_DIM;
  float sse = 0.f;
  {
    const int row = tid;
    const float* e = cbl32 + (size_t)idxS[row] * D_DIM;
    #pragma unroll 4
    for (int d = 0; d < D_DIM; ++d) {
      size_t off = (size_t)d * T_LEN + row;
      float r = Rblk[off] - e[d];
      Rblk[off] = r;
      sse += r * r;
    }
  }
  #pragma unroll
  for (int off = 16; off > 0; off >>= 1) sse += __shfl_xor(sse, off, 32);
  if (lane == 0) atomicAdd(loss_acc, sse);
}

// ---------------- finish: quantized_out = xin - residual_final; write loss ----------------
__global__ void finish_kernel(const float4* __restrict__ xin, float4* __restrict__ q,
                              const float* __restrict__ loss_acc, float* __restrict__ loss_out,
                              size_t n4) {
  size_t i = (size_t)blockIdx.x * blockDim.x + threadIdx.x;
  size_t stride = (size_t)gridDim.x * blockDim.x;
  for (; i < n4; i += stride) {
    float4 a = xin[i], r = q[i];
    q[i] = make_float4(a.x - r.x, a.y - r.y, a.z - r.z, a.w - r.w);
  }
  if (blockIdx.x == 0 && threadIdx.x == 0) {
    // total_loss = mean_l( 1.25 * SSE_l / (B*T*C) ) = 1.25 * SSE_total / (L * B*T*C)
    *loss_out = (1.25f / ((float)L_LAYERS * 33554432.0f)) * (*loss_acc);
  }
}

extern "C" void kernel_launch(void* const* d_in, const int* in_sizes, int n_in,
                              void* d_out, int out_size, void* d_ws, size_t ws_size,
                              hipStream_t stream) {
  (void)in_sizes; (void)n_in; (void)out_size; (void)ws_size;
  const float* xin = (const float*)d_in[0];     // [B,C,T]
  const float* cb  = (const float*)d_in[1];     // [L,G,K,D]

  float* out      = (float*)d_out;
  float* q        = out;                         // residual lives here during the layer loop
  float* loss_out = out + QN;
  float* idx_out  = out + QN + 1;                // [L,G,N] float-encoded

  // workspace: bf16 codebook (8 MB) + norms (64 KB) + loss accumulator (4 B)
  const size_t CBE = (size_t)L_LAYERS * G_GRP * K_CODES * D_DIM;   // 4,194,304 elements
  unsigned short* cbbf  = (unsigned short*)d_ws;
  float*          norms = (float*)((char*)d_ws + CBE * sizeof(unsigned short));
  float*          lacc  = norms + (size_t)L_LAYERS * G_GRP * K_CODES;

  hipMemsetAsync(lacc, 0, sizeof(float), stream);

  {
    int rows = L_LAYERS * G_GRP * K_CODES;
    prep_codebook<<<(rows + 255) / 256, 256, 0, stream>>>(cb, cbbf, norms);
  }
  copy_kernel<<<2048, 256, 0, stream>>>((const float4*)xin, (float4*)q, QN / 4);

  for (int l = 0; l < L_LAYERS; ++l) {
    rvq_layer<<<dim3(N_VEC / 64, G_GRP), 64, 0, stream>>>(
        q, cb, cbbf, norms,
        idx_out + (size_t)l * G_GRP * N_VEC, lacc, l);
  }

  finish_kernel<<<2048, 256, 0, stream>>>((const float4*)xin, (float4*)q, lacc, loss_out, QN / 4);
}
